// MultiLayerCNN2D_41412074668084
// MI455X (gfx1250) — matrix operand order repaired and checked by reading, not compiled, not run on hardware
//
#include <hip/hip_runtime.h>
#include <hip/hip_bf16.h>
#include <math.h>

// ---------------------------------------------------------------------------
// Types for CDNA5 WMMA (wave32): v_wmma_f32_16x16x32_bf16
// ---------------------------------------------------------------------------
typedef __attribute__((ext_vector_type(16))) __bf16        v16bf;
typedef __attribute__((ext_vector_type(8)))  float         v8f;
typedef __attribute__((ext_vector_type(8)))  unsigned short ushort8;

union Frag16 {
    v16bf   bf;
    ushort8 u8[2];   // two 16-byte halves
};

static __device__ __forceinline__ unsigned short f32_to_bf16_rne(float f) {
    unsigned int u = __builtin_bit_cast(unsigned int, f);
    u += 0x7FFFu + ((u >> 16) & 1u);           // round-to-nearest-even
    return (unsigned short)(u >> 16);
}

static __device__ __forceinline__ v8f wmma_bf16(const Frag16& a, const Frag16& b, v8f c) {
    return __builtin_amdgcn_wmma_f32_16x16x32_bf16(
        /*neg_a=*/false, a.bf, /*neg_b=*/false, b.bf,
        /*c_mod=*/(short)0, c, /*reuse_a=*/false, /*reuse_b=*/false);
}

// ---------------------------------------------------------------------------
// Problem constants
// ---------------------------------------------------------------------------
#define BATCH 64
#define CIN   3
#define H0    128
#define C1    64
#define H1SZ  64      // after pool1
#define C2    128
#define H2SZ  32      // after pool2
#define FC    256
#define NCLS  1000
#define FLAT  (C2*H2SZ*H2SZ)   // 131072

// ===========================================================================
// Kernel 1: conv1 (3->64, 3x3, pad1) + ReLU + 2x2 maxpool, output bf16
//   One thread per pooled output element: [64,64,64,64]
// ===========================================================================
__global__ __launch_bounds__(256)
void conv1_relu_pool_kernel(const float* __restrict__ x,      // [64,3,128,128]
                            const float* __restrict__ w,      // [64,27]
                            const float* __restrict__ bias,   // [64]
                            unsigned short* __restrict__ h1)  // [64,64,64,64] bf16
{
    const int idx = blockIdx.x * 256 + threadIdx.x;   // 64*64*64*64 = 16.7M
    const int px = idx & 63;
    const int py = (idx >> 6) & 63;
    const int o  = (idx >> 12) & 63;
    const int b  = idx >> 18;

    float wr[27];
#pragma unroll
    for (int i = 0; i < 27; ++i) wr[i] = w[o * 27 + i];
    const float bb = bias[o];

    float best = 0.0f;  // ReLU outputs are >= 0, so 0 is the pooling identity
#pragma unroll
    for (int sy = 0; sy < 2; ++sy) {
#pragma unroll
        for (int sx = 0; sx < 2; ++sx) {
            const int oy = py * 2 + sy;
            const int ox = px * 2 + sx;
            float s = bb;
#pragma unroll
            for (int c = 0; c < CIN; ++c) {
#pragma unroll
                for (int kh = 0; kh < 3; ++kh) {
                    const int iy = oy + kh - 1;
                    if ((unsigned)iy >= (unsigned)H0) continue;
#pragma unroll
                    for (int kw = 0; kw < 3; ++kw) {
                        const int ix = ox + kw - 1;
                        if ((unsigned)ix < (unsigned)H0)
                            s += wr[c * 9 + kh * 3 + kw] *
                                 x[((b * CIN + c) * H0 + iy) * H0 + ix];
                    }
                }
            }
            s = s > 0.0f ? s : 0.0f;
            best = fmaxf(best, s);
        }
    }
    h1[idx] = f32_to_bf16_rne(best);   // layout (b,o,py,px) == idx decomposition
}

// ===========================================================================
// Kernel 2: conv2 (64->128, 3x3, pad1) + ReLU + 2x2 maxpool via WMMA
//   Implicit GEMM: M = 64 pixels (8x8 spatial block), N = 128, K = 576
//   K factored as 9 taps x 2 channel-halves of 32.
//   Block = 256 threads = 8 waves; wave w owns N columns [16w, 16w+16).
//   All 5 fragments are loaded before the 4 WMMAs so DS latency overlaps
//   the XDL pipeline instead of serializing on s_wait_dscnt 0.
// ===========================================================================
__global__ __launch_bounds__(256)
void conv2_wmma_kernel(const unsigned short* __restrict__ h1,  // [64,64,64,64] bf16
                       const float* __restrict__ w2,           // [128,576]
                       const float* __restrict__ b2,           // [128]
                       unsigned short* __restrict__ h2)        // [64,128,32,32] bf16
{
    __shared__ unsigned short a_tile[64][40];    // 64 pixels x 32 K (+pad, 16B-aligned rows)
    __shared__ unsigned short b_tile[128][40];   // stored [n][k]
    __shared__ float          out_tile[64][132]; // pre-pool conv outputs (pixel, channel)

    const int tid  = threadIdx.x;
    const int wave = tid >> 5;
    const int lane = tid & 31;

    const int blk = blockIdx.x;          // 64 * 8 * 8
    const int b   = blk >> 6;
    const int by  = (blk >> 3) & 7;      // spatial block row (of 8x8 pixels)
    const int bx  = blk & 7;

    const int n16   = wave * 16;
    const int mrow  = lane & 15;
    const int hi    = lane >> 4;
    const int kb    = hi * 8;            // A-fragment K base per lane half
    const int khalf = hi;                // B-fragment K half per lane half

    v8f acc[4];
#pragma unroll
    for (int t = 0; t < 4; ++t)
#pragma unroll
        for (int r = 0; r < 8; ++r) acc[t][r] = 0.0f;

    for (int kk = 0; kk < 9; ++kk) {
        const int kh = kk / 3, kw = kk % 3;
        for (int ch = 0; ch < 2; ++ch) {
            const int c0 = ch * 32;
            __syncthreads();   // previous iteration's reads done before overwrite

            // ---- stage A: im2col 64 pixels x 32 channels into LDS ----
            for (int e = tid; e < 64 * 32; e += 256) {
                const int m  = e >> 5;
                const int kc = e & 31;
                const int iy = by * 8 + (m >> 3) + kh - 1;
                const int ix = bx * 8 + (m & 7) + kw - 1;
                unsigned short v = 0;
                if ((unsigned)iy < (unsigned)H1SZ && (unsigned)ix < (unsigned)H1SZ)
                    v = h1[((b * C1 + (c0 + kc)) * H1SZ + iy) * H1SZ + ix];
                a_tile[m][kc] = v;
            }
            // ---- stage B: weights, f32 -> bf16, stored [n][k] ----
            for (int e = tid; e < 128 * 32; e += 256) {
                const int n  = e >> 5;
                const int kc = e & 31;
                b_tile[n][kc] =
                    f32_to_bf16_rne(w2[n * 576 + (c0 + kc) * 9 + kh * 3 + kw]);
            }
            __syncthreads();

            // ---- load ALL fragments first, then issue 4 independent WMMAs ----
            Frag16 fb, fa[4];
            fb.u8[0] = *(const ushort8*)&b_tile[n16 + mrow][khalf * 8];
            fb.u8[1] = *(const ushort8*)&b_tile[n16 + mrow][khalf * 8 + 16];
#pragma unroll
            for (int t = 0; t < 4; ++t) {
                fa[t].u8[0] = *(const ushort8*)&a_tile[t * 16 + mrow][kb];
                fa[t].u8[1] = *(const ushort8*)&a_tile[t * 16 + mrow][kb + 16];
            }
#pragma unroll
            for (int t = 0; t < 4; ++t)
                acc[t] = wmma_bf16(fa[t], fb, acc[t]);
        }
    }

    // ---- epilogue: bias + ReLU into LDS, then 2x2 pool -> bf16 global ----
    __syncthreads();
    const float bias = b2[n16 + mrow];
#pragma unroll
    for (int t = 0; t < 4; ++t)
#pragma unroll
        for (int r = 0; r < 8; ++r) {
            const float v = acc[t][r] + bias;
            out_tile[t * 16 + r + 8 * hi][n16 + mrow] = v > 0.0f ? v : 0.0f;
        }
    __syncthreads();

    for (int e = tid; e < 16 * 128; e += 256) {
        const int p  = e >> 7;           // pooled pixel 0..15 (4x4)
        const int n  = e & 127;
        const int py = p >> 2, px = p & 3;
        const int m0 = (py * 2) * 8 + px * 2;   // top-left pixel in 8x8 block
        const float mx = fmaxf(fmaxf(out_tile[m0][n],     out_tile[m0 + 1][n]),
                               fmaxf(out_tile[m0 + 8][n], out_tile[m0 + 9][n]));
        const int oy = by * 4 + py, ox = bx * 4 + px;
        h2[((b * C2 + n) * H2SZ + oy) * H2SZ + ox] = f32_to_bf16_rne(mx);
    }
}

// ===========================================================================
// Kernel 3: fc1  out[64,256] = relu(A[64,131072] @ W[256,131072]^T + b), bf16 out
//   16 blocks (one 16-wide N tile each), 8 waves split K (16384 each),
//   partials reduced through LDS. Weights streamed once (prefetch ahead);
//   activations are bf16 and L2-resident across the 16 N-tiles.
// ===========================================================================
__global__ __launch_bounds__(256)
void fc1_wmma_kernel(const unsigned short* __restrict__ A,     // [64,131072] bf16
                     const float* __restrict__ W,              // [256,131072]
                     const float* __restrict__ bias,           // [256]
                     unsigned short* __restrict__ out)         // [64,256] bf16
{
    __shared__ float red[8][4][16][16];   // wave x m-tile x 16 x 16 = 32 KB

    const int tid  = threadIdx.x;
    const int wave = tid >> 5;
    const int lane = tid & 31;
    const int n0   = blockIdx.x * 16;
    const int mrow = lane & 15;
    const int hi   = lane >> 4;
    const int kb   = hi * 8;
    const int khalf = hi;

    v8f acc[4];
#pragma unroll
    for (int t = 0; t < 4; ++t)
#pragma unroll
        for (int r = 0; r < 8; ++r) acc[t][r] = 0.0f;

    const size_t KTOT   = (size_t)FLAT;
    const size_t kSlice = KTOT / 8;
    const size_t kBegin = (size_t)wave * kSlice;
    const float* wrow   = W + (size_t)(n0 + mrow) * KTOT;

    for (size_t k = kBegin; k < kBegin + kSlice; k += 32) {
        const float* wp = wrow + k + (size_t)khalf * 8;
        __builtin_prefetch(wp + 1024, 0, 1);      // global_prefetch ~32 steps ahead

        // load all A fragments before converting B / issuing WMMAs
        Frag16 fa[4];
#pragma unroll
        for (int t = 0; t < 4; ++t) {
            const unsigned short* ap = A + (size_t)(t * 16 + mrow) * KTOT + k;
            fa[t].u8[0] = *(const ushort8*)(ap + kb);
            fa[t].u8[1] = *(const ushort8*)(ap + kb + 16);
        }
        Frag16 fb;
#pragma unroll
        for (int j = 0; j < 8; ++j) {
            fb.u8[0][j] = f32_to_bf16_rne(wp[j]);
            fb.u8[1][j] = f32_to_bf16_rne(wp[16 + j]);
        }
#pragma unroll
        for (int t = 0; t < 4; ++t)
            acc[t] = wmma_bf16(fa[t], fb, acc[t]);
    }

    // partials -> LDS
#pragma unroll
    for (int t = 0; t < 4; ++t)
#pragma unroll
        for (int r = 0; r < 8; ++r)
            red[wave][t][r + 8 * hi][mrow] = acc[t][r];
    __syncthreads();

    // cross-wave reduce + bias + ReLU -> bf16
    for (int e = tid; e < 64 * 16; e += 256) {
        const int m  = e >> 4;           // 0..63
        const int nl = e & 15;
        const int t  = m >> 4, ml = m & 15;
        float s = 0.0f;
#pragma unroll
        for (int w = 0; w < 8; ++w) s += red[w][t][ml][nl];
        s += bias[n0 + nl];
        s = s > 0.0f ? s : 0.0f;
        out[m * FC + n0 + nl] = f32_to_bf16_rne(s);
    }
}

// ===========================================================================
// Kernel 4: fc2  out[64,1000] = A[64,256] @ W[1000,256]^T + b  (f32 out)
//   63 blocks x 4 waves; wave w owns M tile w; N guarded at 1000.
// ===========================================================================
__global__ __launch_bounds__(128)
void fc2_wmma_kernel(const unsigned short* __restrict__ A,   // [64,256] bf16
                     const float* __restrict__ W,            // [1000,256]
                     const float* __restrict__ bias,         // [1000]
                     float* __restrict__ out)                // [64,1000]
{
    const int tid  = threadIdx.x;
    const int wave = tid >> 5;
    const int lane = tid & 31;
    const int n0   = blockIdx.x * 16;
    const int mrow = lane & 15;
    const int hi   = lane >> 4;
    const int kb   = hi * 8;
    const int khalf = hi;
    const int m0   = wave * 16;

    const int n  = n0 + mrow;
    const int nc = (n < NCLS) ? n : 0;

    v8f acc;
#pragma unroll
    for (int r = 0; r < 8; ++r) acc[r] = 0.0f;

    const float* wrow = W + (size_t)nc * FC;
#pragma unroll
    for (int k = 0; k < FC; k += 32) {
        Frag16 fa, fb;
        const unsigned short* ap = A + (size_t)(m0 + mrow) * FC + k;
        fa.u8[0] = *(const ushort8*)(ap + kb);
        fa.u8[1] = *(const ushort8*)(ap + kb + 16);
        const float* wp = wrow + k + khalf * 8;
#pragma unroll
        for (int j = 0; j < 8; ++j) {
            fb.u8[0][j] = f32_to_bf16_rne(wp[j]);
            fb.u8[1][j] = f32_to_bf16_rne(wp[16 + j]);
        }
        acc = wmma_bf16(fa, fb, acc);
    }

    if (n < NCLS) {
        const float bv = bias[n];
#pragma unroll
        for (int r = 0; r < 8; ++r) {
            const int m = m0 + r + 8 * hi;
            out[(size_t)m * NCLS + n] = acc[r] + bv;
        }
    }
}

// ===========================================================================
// Host launch
// ===========================================================================
extern "C" void kernel_launch(void* const* d_in, const int* in_sizes, int n_in,
                              void* d_out, int out_size, void* d_ws, size_t ws_size,
                              hipStream_t stream) {
    (void)in_sizes; (void)n_in; (void)out_size; (void)ws_size;

    const float* x   = (const float*)d_in[0];   // [64,3,128,128]
    const float* w1  = (const float*)d_in[1];   // [64,27]
    const float* b1  = (const float*)d_in[2];   // [64]
    const float* w2  = (const float*)d_in[3];   // [128,576]
    const float* b2  = (const float*)d_in[4];   // [128]
    const float* w5  = (const float*)d_in[5];   // [256,131072]
    const float* b5  = (const float*)d_in[6];   // [256]
    const float* w7  = (const float*)d_in[7];   // [1000,256]
    const float* b7  = (const float*)d_in[8];   // [1000]
    float* y         = (float*)d_out;           // [64,1000]

    // Workspace layout (bf16 intermediates)
    char* ws = (char*)d_ws;
    unsigned short* h1 = (unsigned short*)(ws);                              // 33.6 MB
    unsigned short* h2 = (unsigned short*)(ws + (size_t)BATCH*C1*H1SZ*H1SZ*2);   // 16.8 MB
    unsigned short* f1 = (unsigned short*)(ws + (size_t)BATCH*C1*H1SZ*H1SZ*2
                                              + (size_t)BATCH*C2*H2SZ*H2SZ*2);  // 32 KB

    // conv1 + relu + pool : one thread per pooled output
    conv1_relu_pool_kernel<<< (BATCH*C1*H1SZ*H1SZ)/256, 256, 0, stream >>>(x, w1, b1, h1);

    // conv2 + relu + pool : implicit GEMM on WMMA
    conv2_wmma_kernel<<< BATCH * 8 * 8, 256, 0, stream >>>(h1, w2, b2, h2);

    // fc1 + relu : K-split WMMA GEMM
    fc1_wmma_kernel<<< FC / 16, 256, 0, stream >>>(h2, w5, b5, f1);

    // fc2 : WMMA GEMM, f32 output
    fc2_wmma_kernel<<< (NCLS + 15) / 16, 128, 0, stream >>>(f1, w7, b7, y);
}
